// Block6StructureFactorEncoder_50517405335836
// MI455X (gfx1250) — matrix operand order repaired and checked by reading, not compiled, not run on hardware
//
#include <hip/hip_runtime.h>
#include <hip/hip_bf16.h>
#include <math.h>

typedef _Float16 h16;
typedef __attribute__((ext_vector_type(8)))  _Float16 v8h;
typedef __attribute__((ext_vector_type(16))) _Float16 v16h;
typedef __attribute__((ext_vector_type(8)))  float    v8f;

#define CAT16(lo, hi) __builtin_shufflevector(lo, hi, 0,1,2,3,4,5,6,7,8,9,10,11,12,13,14,15)
#define WMMA(a, b, c) __builtin_amdgcn_wmma_f32_16x16x32_f16(false, a, false, b, (short)0, c, false, false)

// ===========================================================================
// Dense GEMM, f16 in / f32 accumulate, 32x32 tile per wave (2x2 register
// blocking, 4 independent accumulators: 4 v_wmma per 8 b128 loads, 16 FLOP/B).
//   A: f16 row-major (M x Kpad), M % 32 == 0, Kpad % 32 == 0, fully readable.
//   Bp: f16 pre-packed per-(ntile,kstep) 512-element WMMA-B fragment blocks,
//       zero-padded -> NO bounds checks in the main loop.
// Tile ids are wave-uniform -> EXEC all-ones at every wmma (ISA requirement).
// ===========================================================================
__global__ __launch_bounds__(256) void gemm_f16(
    const h16* __restrict__ A, int lda,
    const h16* __restrict__ Bp,
    const float* __restrict__ bias,
    float* __restrict__ C32, int ldc,
    h16* __restrict__ C16, int ldc16,
    int M, int N, int Kpad, int act)
{
  const int lane = threadIdx.x & 31;
  const int wave = threadIdx.x >> 5;
  const int half = lane >> 4, l16 = lane & 15;
  const int ntiles = (N + 15) >> 4;
  const int nt2 = (ntiles + 1) >> 1;
  const int mt2 = M >> 5;
  const int tile = blockIdx.x * 8 + wave;
  if (tile >= nt2 * mt2) return;
  const int tn = tile % nt2;
  const int tm = tile / nt2;
  const int ksteps = Kpad >> 5;
  const bool n1ok = (tn * 2 + 1) < ntiles;
  const int tn1 = n1ok ? (tn * 2 + 1) : 0;   // clamp: data finite, store skipped

  const h16* ap0 = A + (size_t)(tm * 32 + l16) * lda + half * 8;
  const h16* ap1 = ap0 + (size_t)16 * lda;
  const h16* bp0 = Bp + (size_t)(tn * 2) * ksteps * 512 + lane * 16;
  const h16* bp1 = Bp + (size_t)tn1 * ksteps * 512 + lane * 16;

  v8f c00 = {}, c01 = {}, c10 = {}, c11 = {};
#pragma unroll 2
  for (int ks = 0; ks < ksteps; ++ks) {
    v8h a0l = *(const v8h*)(ap0);
    v8h a0h = *(const v8h*)(ap0 + 16);
    v8h a1l = *(const v8h*)(ap1);
    v8h a1h = *(const v8h*)(ap1 + 16);
    v8h b0l = *(const v8h*)(bp0);
    v8h b0h = *(const v8h*)(bp0 + 8);
    v8h b1l = *(const v8h*)(bp1);
    v8h b1h = *(const v8h*)(bp1 + 8);
    v16h a0 = CAT16(a0l, a0h);
    v16h a1 = CAT16(a1l, a1h);
    v16h b0 = CAT16(b0l, b0h);
    v16h b1 = CAT16(b1l, b1h);
    c00 = WMMA(a0, b0, c00);
    c01 = WMMA(a0, b1, c01);
    c10 = WMMA(a1, b0, c10);
    c11 = WMMA(a1, b1, c11);
    ap0 += 32; ap1 += 32; bp0 += 512; bp1 += 512;
  }

#pragma unroll
  for (int ni = 0; ni < 2; ++ni) {
    const int tnn = tn * 2 + ni;
    if (tnn >= ntiles) continue;             // wave-uniform
    const int ncol = tnn * 16 + l16;
    const float bv_ = (bias && ncol < N) ? bias[ncol] : 0.f;
#pragma unroll
    for (int mi = 0; mi < 2; ++mi) {
      const v8f acc = (mi == 0) ? (ni == 0 ? c00 : c01) : (ni == 0 ? c10 : c11);
#pragma unroll
      for (int v = 0; v < 8; ++v) {
        const int r = tm * 32 + mi * 16 + half * 8 + v;  // C/D: M = vgpr+8*half
        float val = acc[v] + bv_;
        if (act == 1) val = fmaxf(val, 0.f);
        if (C32 && ncol < N) C32[(size_t)r * ldc + ncol] = val;
        if (C16) C16[(size_t)r * ldc16 + ncol] = (h16)val;  // pad cols exact 0
      }
    }
  }
}

// Same 2x2 main loop, conv epilogue: BN+ReLU, f16 NCHW (nb, cout, 92, 59).
// cout in {32,64,128}: ntiles even, M % 32 == 0 -> no guards at all.
__global__ __launch_bounds__(256) void gemm_f16_conv(
    const h16* __restrict__ A, int lda,
    const h16* __restrict__ Bp,
    const float* __restrict__ cb,  const float* __restrict__ bng,
    const float* __restrict__ bnb, const float* __restrict__ bnm,
    const float* __restrict__ bnv,
    h16* __restrict__ out, int cout, int M, int Kpad)
{
  const int lane = threadIdx.x & 31;
  const int wave = threadIdx.x >> 5;
  const int half = lane >> 4, l16 = lane & 15;
  const int nt2 = cout >> 5;
  const int mt2 = M >> 5;
  const int tile = blockIdx.x * 8 + wave;
  if (tile >= nt2 * mt2) return;
  const int tn = tile % nt2;
  const int tm = tile / nt2;
  const int ksteps = Kpad >> 5;

  const h16* ap0 = A + (size_t)(tm * 32 + l16) * lda + half * 8;
  const h16* ap1 = ap0 + (size_t)16 * lda;
  const h16* bp0 = Bp + (size_t)(tn * 2) * ksteps * 512 + lane * 16;
  const h16* bp1 = bp0 + (size_t)ksteps * 512;

  v8f c00 = {}, c01 = {}, c10 = {}, c11 = {};
#pragma unroll 2
  for (int ks = 0; ks < ksteps; ++ks) {
    v8h a0l = *(const v8h*)(ap0);
    v8h a0h = *(const v8h*)(ap0 + 16);
    v8h a1l = *(const v8h*)(ap1);
    v8h a1h = *(const v8h*)(ap1 + 16);
    v8h b0l = *(const v8h*)(bp0);
    v8h b0h = *(const v8h*)(bp0 + 8);
    v8h b1l = *(const v8h*)(bp1);
    v8h b1h = *(const v8h*)(bp1 + 8);
    v16h a0 = CAT16(a0l, a0h);
    v16h a1 = CAT16(a1l, a1h);
    v16h b0 = CAT16(b0l, b0h);
    v16h b1 = CAT16(b1l, b1h);
    c00 = WMMA(a0, b0, c00);
    c01 = WMMA(a0, b1, c01);
    c10 = WMMA(a1, b0, c10);
    c11 = WMMA(a1, b1, c11);
    ap0 += 32; ap1 += 32; bp0 += 512; bp1 += 512;
  }

#pragma unroll
  for (int ni = 0; ni < 2; ++ni) {
    const int ncol = (tn * 2 + ni) * 16 + l16;
    const float scale = bng[ncol] / sqrtf(bnv[ncol] + 1e-5f);
    const float shift = (cb[ncol] - bnm[ncol]) * scale + bnb[ncol];
#pragma unroll
    for (int mi = 0; mi < 2; ++mi) {
      const v8f acc = (mi == 0) ? (ni == 0 ? c00 : c01) : (ni == 0 ? c10 : c11);
#pragma unroll
      for (int v = 0; v < 8; ++v) {
        const int r = tm * 32 + mi * 16 + half * 8 + v;
        const int bo = r / 5428;             // constant divisor -> mul/shift
        const int re = r - bo * 5428;
        out[((size_t)bo * cout + ncol) * 5428 + re] =
            (h16)fmaxf(acc[v] * scale + shift, 0.f);
      }
    }
  }
}

// ===========================================================================
// Weight pre-pack: f32 -> f16 WMMA-B fragment order, zero padded.
// transposed==0: W is (K x N) row-major. transposed==1: W is (N x K) (OIHW).
// ===========================================================================
__global__ void pack_w(const float* __restrict__ W, h16* __restrict__ out,
                       int K, int N, int ksteps, int transposed, long total)
{
  const long e = (long)blockIdx.x * 256 + threadIdx.x;
  if (e >= total) return;
  const int i = (int)(e & 15);
  const int lane = (int)((e >> 4) & 31);
  const long blk = e >> 9;
  const int ks = (int)(blk % ksteps);
  const int tn = (int)(blk / ksteps);
  const int k = ks * 32 + (lane >> 4) * 8 + (i & 7) + (i >> 3) * 16;
  const int n = tn * 16 + (lane & 15);
  float v = 0.f;
  if (k < K && n < N)
    v = transposed ? W[(size_t)n * K + k] : W[(size_t)k * N + n];
  out[e] = (h16)v;
}

// sf[:, :, 4:63] -> f16 image (256, 1, 92, 59)
__global__ void recip_to_h16(const float* __restrict__ sf, h16* __restrict__ out)
{
  const int i = blockIdx.x * 256 + threadIdx.x;
  if (i >= 256 * 5428) return;
  const int b = i / 5428, rem = i - b * 5428;
  const int y = rem / 59, x = rem - (rem / 59) * 59;
  out[i] = (h16)sf[((size_t)b * 92 + y) * 63 + 4 + x];
}

// im2col: f16 NCHW -> f16 (nb*5428 x KPAD) row-major, zero padded/bordered.
template <int KSZ, int KPAD, int CIN>
__global__ __launch_bounds__(256) void im2col_k(
    const h16* __restrict__ in, h16* __restrict__ out, int nb)
{
  constexpr int KK = KSZ * KSZ, PAD = KSZ / 2, K = CIN * KK, P = 5428;
  const long e = (long)blockIdx.x * 256 + threadIdx.x;
  if (e >= (long)nb * P * KPAD) return;
  const int k = (int)(e % KPAD);
  const long m = e / KPAD;
  h16 v = (h16)0.f;
  if (k < K) {
    const int bb = (int)(m / P);
    const int rem = (int)(m - (long)bb * P);
    const int y = rem / 59, x = rem - (rem / 59) * 59;
    const int ci = k / KK, r = k - ci * KK;
    const int ky = r / KSZ, kx = r - (r / KSZ) * KSZ;
    const int yy = y + ky - PAD, xx = x + kx - PAD;
    if (yy >= 0 && yy < 92 && xx >= 0 && xx < 59)
      v = in[((long)bb * CIN + ci) * P + yy * 59 + xx];
  }
  out[e] = v;
}

// Pack variant-MLP inputs from sf[:, :, 0:4] into padded f16 rows.
__global__ void pack_real16(const float* __restrict__ sf,
                            h16* a0, h16* a1, h16* a2)
{
  const int b = blockIdx.x;
  for (int j = threadIdx.x; j < 192; j += 256)
    a0[(size_t)b * 192 + j] =
        (h16)(j < 184 ? sf[((size_t)b * 92 + j / 2) * 63 + (j % 2)] : 0.f);
  for (int j = threadIdx.x; j < 288; j += 256)
    a1[(size_t)b * 288 + j] =
        (h16)(j < 276 ? sf[((size_t)b * 92 + j / 3) * 63 + (j % 3)] : 0.f);
  for (int j = threadIdx.x; j < 384; j += 256)
    a2[(size_t)b * 384 + j] =
        (h16)(j < 368 ? sf[((size_t)b * 92 + j / 4) * 63 + (j % 4)] : 0.f);
}

// n = #cols c<4 with sum_h |sf[b,h,c]| > 1e-6; select variant row -> f32+f16
__global__ void select_rf(const float* __restrict__ sf,
                          const float* __restrict__ o0,
                          const float* __restrict__ o1,
                          const float* __restrict__ o2,
                          float* __restrict__ rf32, h16* __restrict__ rf16)
{
  const int b = blockIdx.x;
  __shared__ float cs[4];
  if (threadIdx.x < 4) {
    float s = 0.f;
    for (int h = 0; h < 92; ++h)
      s += fabsf(sf[((size_t)b * 92 + h) * 63 + threadIdx.x]);
    cs[threadIdx.x] = s;
  }
  __syncthreads();
  int n = 0;
#pragma unroll
  for (int c = 0; c < 4; ++c) n += (cs[c] > 1e-6f) ? 1 : 0;
  const float* src = (n == 2) ? o0 : ((n == 3) ? o1 : o2);
  const float v = src[(size_t)b * 256 + threadIdx.x];
  rf32[(size_t)b * 256 + threadIdx.x] = v;
  rf16[(size_t)b * 256 + threadIdx.x] = (h16)v;
}

// cnn16[b0+bb, c, o, p] = window mean of x3_16[bb, c, :, :]   (f32 accumulate)
__global__ void pool_kernel(const h16* __restrict__ x3,
                            h16* __restrict__ cnn, int nb, int b0)
{
  const int idx = blockIdx.x * blockDim.x + threadIdx.x;
  if (idx >= nb * 128 * 64) return;
  const int p = idx & 7, o = (idx >> 3) & 7;
  const int c = (idx >> 6) % 128;
  const int bb = idx / (128 * 64);
  const int hs = o * 92 / 8, he = ((o + 1) * 92 + 7) / 8;
  const int ws = p * 59 / 8, we = ((p + 1) * 59 + 7) / 8;
  float s = 0.f;
  for (int h = hs; h < he; ++h)
    for (int w = ws; w < we; ++w)
      s += (float)x3[(((long)bb * 128 + c) * 92 + h) * 59 + w];
  cnn[(((long)(b0 + bb) * 128 + c) * 64) + o * 8 + p] =
      (h16)(s / (float)((he - hs) * (we - ws)));
}

// kv16[b, c] = mean_{o,p} cnn[b,c,o,p] for c<128 else 0 (zero-pad to 256)
__global__ void kv_kernel(const h16* __restrict__ cnn, h16* __restrict__ kv)
{
  const int i = blockIdx.x * blockDim.x + threadIdx.x;
  if (i >= 256 * 256) return;
  const int b = i >> 8, c = i & 255;
  float r = 0.f;
  if (c < 128) {
    const h16* p = cnn + ((size_t)b * 128 + c) * 64;
    float s = 0.f;
    for (int j = 0; j < 64; ++j) s += (float)p[j];
    r = s * (1.f / 64.f);
  }
  kv[i] = (h16)r;
}

// LayerNorm(x + res)*g + b  -> optional f32 and f16 outputs. Wave per row.
__global__ __launch_bounds__(256) void add_ln_kernel(
    const float* __restrict__ x, const float* __restrict__ res,
    const float* __restrict__ g, const float* __restrict__ bta,
    float* __restrict__ out32, h16* __restrict__ out16)
{
  const int wave = threadIdx.x >> 5, lane = threadIdx.x & 31;
  const int row = blockIdx.x * 8 + wave;
  float v[8];
  float s = 0.f;
#pragma unroll
  for (int i = 0; i < 8; ++i) {
    const int c = lane + i * 32;
    v[i] = x[(size_t)row * 256 + c] + res[(size_t)row * 256 + c];
    s += v[i];
  }
  for (int off = 16; off; off >>= 1) s += __shfl_xor(s, off, 32);
  const float mean = s * (1.f / 256.f);
  float vs = 0.f;
#pragma unroll
  for (int i = 0; i < 8; ++i) { const float d = v[i] - mean; vs += d * d; }
  for (int off = 16; off; off >>= 1) vs += __shfl_xor(vs, off, 32);
  const float inv = 1.f / sqrtf(vs * (1.f / 256.f) + 1e-5f);
#pragma unroll
  for (int i = 0; i < 8; ++i) {
    const int c = lane + i * 32;
    const float o = (v[i] - mean) * inv * g[c] + bta[c];
    if (out32) out32[(size_t)row * 256 + c] = o;
    if (out16) out16[(size_t)row * 256 + c] = (h16)o;
  }
}

// ===========================================================================
static inline int kpad32(int K) { return (K + 31) & ~31; }

static inline void gemm(hipStream_t s, const h16* A, int lda, const h16* Bp,
                        const float* bias, float* C32, int ldc,
                        h16* C16, int ldc16, int M, int N, int Kpad, int act)
{
  const int nt2 = (((N + 15) >> 4) + 1) >> 1;
  const int tiles = (M >> 5) * nt2;
  gemm_f16<<<(tiles + 7) / 8, 256, 0, s>>>(A, lda, Bp, bias, C32, ldc,
                                           C16, ldc16, M, N, Kpad, act);
}

extern "C" void kernel_launch(void* const* d_in, const int* in_sizes, int n_in,
                              void* d_out, int out_size, void* d_ws, size_t ws_size,
                              hipStream_t stream)
{
  (void)in_sizes; (void)n_in; (void)out_size;
  const float* sf = (const float*)d_in[0];
  auto P = [&](int i) { return (const float*)d_in[i]; };
  const float *a_w1[3] = {P(1), P(5), P(9)},  *a_b1[3] = {P(2), P(6), P(10)};
  const float *a_w2[3] = {P(3), P(7), P(11)}, *a_b2[3] = {P(4), P(8), P(12)};
  const float *c_w[3]  = {P(13), P(19), P(25)}, *c_b[3]  = {P(14), P(20), P(26)};
  const float *c_bg[3] = {P(15), P(21), P(27)}, *c_bb[3] = {P(16), P(22), P(28)};
  const float *c_bm[3] = {P(17), P(23), P(29)}, *c_bv[3] = {P(18), P(24), P(30)};
  const float *att_wv = P(35), *att_bv = P(36), *att_wo = P(37), *att_bo = P(38);
  const float *ln1_g = P(39), *ln1_b = P(40), *ln2_g = P(41), *ln2_b = P(42);
  const float *ff_w1 = P(43), *ff_b1 = P(44), *ff_w2 = P(45), *ff_b2 = P(46);
  const float *kp_w1 = P(47), *kp_b1 = P(48), *kp_w2 = P(49), *kp_b2 = P(50);
  const float *rr_w1 = P(51), *rr_b1 = P(52), *rr_w2 = P(53), *rr_b2 = P(54);
  const float *er_w1 = P(55), *er_b1 = P(56), *er_w2 = P(57), *er_b2 = P(58);
  const float *rs_w1 = P(59), *rs_b1 = P(60), *rs_w2 = P(61), *rs_b2 = P(62);
  const float *rp_w1 = P(63), *rp_b1 = P(64), *rp_w2 = P(65), *rp_b2 = P(66);
  const float *ip_w1 = P(67), *ip_b1 = P(68), *ip_w2 = P(69), *ip_b2 = P(70);

  // ---- byte-based workspace allocator (256B aligned) ----
  char* base = (char*)d_ws;
  size_t off = 0;
  auto allocB = [&](size_t bytes) {
    char* p = base + off;
    off += (bytes + 255) & ~(size_t)255;
    return p;
  };
  auto allocF = [&](size_t n) { return (float*)allocB(n * 4); };
  auto allocH = [&](size_t n) { return (h16*)allocB(n * 2); };

  // packed weights (f16)
  auto packW = [&](const float* W, int K, int N, int transposed) {
    const int ks = kpad32(K) / 32, nt = (N + 15) / 16;
    const long total = (long)nt * ks * 512;
    h16* out = allocH((size_t)total);
    pack_w<<<(unsigned)((total + 255) / 256), 256, 0, stream>>>(
        W, out, K, N, ks, transposed, total);
    return out;
  };
  h16 *pw_a1[3], *pw_a2[3];
  const int adin[3] = {184, 276, 368};
  for (int v = 0; v < 3; ++v) {
    pw_a1[v] = packW(a_w1[v], adin[v], 256, 0);
    pw_a2[v] = packW(a_w2[v], 256, 256, 0);
  }
  h16* pw_c1 = packW(c_w[0], 25, 32, 1);    // OIHW == (cout x K) -> transposed
  h16* pw_c2 = packW(c_w[1], 288, 64, 1);
  h16* pw_c3 = packW(c_w[2], 576, 128, 1);
  h16* pw_attv = packW(att_wv, 256, 256, 0);
  h16* pw_atto = packW(att_wo, 256, 256, 0);
  h16* pw_ff1 = packW(ff_w1, 256, 512, 0);
  h16* pw_ff2 = packW(ff_w2, 512, 256, 0);
  h16* pw_kp1 = packW(kp_w1, 256, 128, 0);
  h16* pw_kp2 = packW(kp_w2, 128, 92, 0);
  h16* pw_rr1 = packW(rr_w1, 256, 128, 0);
  h16* pw_rr2 = packW(rr_w2, 128, 4, 0);
  h16* pw_er1 = packW(er_w1, 256, 128, 0);
  h16* pw_er2 = packW(er_w2, 128, 368, 0);
  h16* pw_rs1 = packW(rs_w1, 256, 256, 0);
  h16* pw_rs2 = packW(rs_w2, 256, 256, 0);
  h16* pw_rp1 = packW(rp_w1, 8192, 512, 0);
  h16* pw_rp2 = packW(rp_w2, 512, 256, 0);
  h16* pw_ip1 = packW(ip_w1, 256, 256, 0);
  h16* pw_ip2 = packW(ip_w2, 256, 256, 0);

  // activations
  h16* a0in = allocH(256 * 192);
  h16* a1in = allocH(256 * 288);
  h16* a2in = allocH(256 * 384);
  h16* hbig16 = allocH(256 * 512);          // reused hidden (max 256x512)
  float* out0 = allocF(256 * 256);
  float* out1 = allocF(256 * 256);
  float* out2 = allocF(256 * 256);
  float* rf32 = allocF(256 * 256);
  h16*   rf16 = allocH(256 * 256);
  h16* recip16 = allocH((size_t)256 * 5428);
  h16* cnn16 = allocH((size_t)256 * 128 * 64);
  h16* kv16  = allocH(256 * 256);
  h16* v16_  = allocH(256 * 256);
  float* aob   = allocF(256 * 256);
  float* hbuf32 = allocF(256 * 256);
  h16*   hbuf16 = allocH(256 * 256);
  float* ffo   = allocF(256 * 256);
  h16*   inter16 = allocH(256 * 256);
  const size_t fixed = off;

  // conv chunk (multiple of 8 so conv M % 32 == 0):
  // per-sample f16 = (32+64+128)*5428 acts + 576*5428 im2col
  const size_t perB = ((size_t)224 * 5428 + (size_t)576 * 5428) * 2 + 4096;
  const size_t avail = (ws_size > fixed) ? ws_size - fixed : 0;
  int chunk = 8;
  for (int c = 256; c >= 8; c >>= 1)
    if ((size_t)c * perB <= avail) { chunk = c; break; }
  h16* x1 = allocH((size_t)chunk * 32 * 5428);
  h16* x2 = allocH((size_t)chunk * 64 * 5428);
  h16* x3 = allocH((size_t)chunk * 128 * 5428);
  h16* col = allocH((size_t)chunk * 5428 * 576);

  // ---- variant MLPs + select -> real_features ----
  pack_real16<<<256, 256, 0, stream>>>(sf, a0in, a1in, a2in);
  {
    h16* ain[3] = {a0in, a1in, a2in};
    float* aout[3] = {out0, out1, out2};
    for (int v = 0; v < 3; ++v) {
      gemm(stream, ain[v], kpad32(adin[v]), pw_a1[v], a_b1[v],
           nullptr, 0, hbig16, 256, 256, 256, kpad32(adin[v]), 1);
      gemm(stream, hbig16, 256, pw_a2[v], a_b2[v],
           aout[v], 256, nullptr, 0, 256, 256, 256, 0);
    }
  }
  select_rf<<<256, 256, 0, stream>>>(sf, out0, out1, out2, rf32, rf16);

  // ---- CNN trunk: im2col + clean 2x2 WMMA GEMM per layer, chunked ----
  recip_to_h16<<<(256 * 5428 + 255) / 256, 256, 0, stream>>>(sf, recip16);
  for (int b0 = 0; b0 < 256; b0 += chunk) {
    const int nb = chunk;
    const int M = nb * 5428;
    {
      const long t = (long)M * 32;
      im2col_k<5, 32, 1><<<(unsigned)((t + 255) / 256), 256, 0, stream>>>(
          recip16 + (size_t)b0 * 5428, col, nb);
      const int tiles = (M >> 5) * (32 >> 5);
      gemm_f16_conv<<<(tiles + 7) / 8, 256, 0, stream>>>(
          col, 32, pw_c1, c_b[0], c_bg[0], c_bb[0], c_bm[0], c_bv[0],
          x1, 32, M, 32);
    }
    {
      const long t = (long)M * 288;
      im2col_k<3, 288, 32><<<(unsigned)((t + 255) / 256), 256, 0, stream>>>(
          x1, col, nb);
      const int tiles = (M >> 5) * (64 >> 5);
      gemm_f16_conv<<<(tiles + 7) / 8, 256, 0, stream>>>(
          col, 288, pw_c2, c_b[1], c_bg[1], c_bb[1], c_bm[1], c_bv[1],
          x2, 64, M, 288);
    }
    {
      const long t = (long)M * 576;
      im2col_k<3, 576, 64><<<(unsigned)((t + 255) / 256), 256, 0, stream>>>(
          x2, col, nb);
      const int tiles = (M >> 5) * (128 >> 5);
      gemm_f16_conv<<<(tiles + 7) / 8, 256, 0, stream>>>(
          col, 576, pw_c3, c_b[2], c_bg[2], c_bb[2], c_bm[2], c_bv[2],
          x3, 128, M, 576);
    }
    const int np = nb * 128 * 64;
    pool_kernel<<<(np + 255) / 256, 256, 0, stream>>>(x3, cnn16, nb, b0);
  }
  kv_kernel<<<256, 256, 0, stream>>>(cnn16, kv16);

  // ---- attention: softmax over singleton axis == 1  =>  attn == v ----
  gemm(stream, kv16, 256, pw_attv, att_bv, nullptr, 0, v16_, 256, 256, 256, 256, 0);
  gemm(stream, v16_, 256, pw_atto, att_bo, aob, 256, nullptr, 0, 256, 256, 256, 0);
  add_ln_kernel<<<32, 256, 0, stream>>>(rf32, aob, ln1_g, ln1_b, hbuf32, hbuf16);

  // ---- feed-forward + LN2 ----
  gemm(stream, hbuf16, 256, pw_ff1, ff_b1, nullptr, 0, hbig16, 512, 256, 512, 256, 1);
  gemm(stream, hbig16, 512, pw_ff2, ff_b2, ffo, 256, nullptr, 0, 256, 256, 512, 0);
  add_ln_kernel<<<32, 256, 0, stream>>>(hbuf32, ffo, ln2_g, ln2_b, nullptr, inter16);

  // ---- heads, straight into concatenated d_out ----
  float* outp = (float*)d_out;
  float* o_kp    = outp;                       // 256 x 92
  float* o_rr    = outp + 256 * 92;            // 256 x 4
  float* o_er    = o_rr + 256 * 4;             // 256 x 368
  float* o_feats = o_er + (size_t)256 * 368;   // 256 x 768

  gemm(stream, inter16, 256, pw_kp1, kp_b1, nullptr, 0, hbig16, 128, 256, 128, 256, 1);
  gemm(stream, hbig16, 128, pw_kp2, kp_b2, o_kp, 92, nullptr, 0, 256, 92, 128, 0);

  gemm(stream, inter16, 256, pw_rr1, rr_b1, nullptr, 0, hbig16, 128, 256, 128, 256, 1);
  gemm(stream, hbig16, 128, pw_rr2, rr_b2, o_rr, 4, nullptr, 0, 256, 4, 128, 0);

  gemm(stream, rf16, 256, pw_er1, er_b1, nullptr, 0, hbig16, 128, 256, 128, 256, 1);
  gemm(stream, hbig16, 128, pw_er2, er_b2, o_er, 368, nullptr, 0, 256, 368, 128, 0);

  gemm(stream, rf16, 256, pw_rs1, rs_b1, nullptr, 0, hbig16, 256, 256, 256, 256, 1);
  gemm(stream, hbig16, 256, pw_rs2, rs_b2, o_feats, 768, nullptr, 0, 256, 256, 256, 0);

  gemm(stream, cnn16, 8192, pw_rp1, rp_b1, nullptr, 0, hbig16, 512, 256, 512, 8192, 1);
  gemm(stream, hbig16, 512, pw_rp2, rp_b2, o_feats + 256, 768, nullptr, 0, 256, 256, 512, 0);

  gemm(stream, inter16, 256, pw_ip1, ip_b1, nullptr, 0, hbig16, 256, 256, 256, 256, 1);
  gemm(stream, hbig16, 256, pw_ip2, ip_b2, o_feats + 512, 768, nullptr, 0, 256, 256, 256, 0);
}
